// PerceiverLayer_39737037423015
// MI455X (gfx1250) — matrix operand order repaired and checked by reading, not compiled, not run on hardware
//
#include <hip/hip_runtime.h>
#include <hip/hip_bf16.h>
#include <stdint.h>

// ---------------------------------------------------------------------------
// PerceiverLayer for MI455X (gfx1250, wave32, WMMA).
// All GEMMs (conv-as-GEMM, attention, projections, FFN) run through
// v_wmma_f32_16x16x32_bf16 with f32 accumulate; softmax/LN/BN stay f32.
// Register-staged double buffering overlaps global loads with WMMA.
// Conv folds its 5 taps into one kernel via an in-kernel tap loop.
// ---------------------------------------------------------------------------

typedef __bf16 bf16;
typedef __attribute__((ext_vector_type(16))) __bf16 v16bf;
typedef __attribute__((ext_vector_type(8)))  float  v8f;

union AFrag { v16bf v; bf16 e[16]; };
union CFrag { v8f  v; float e[8]; };
union Pack4 { unsigned long long u; bf16 e[4]; };

struct Taps { long o[5]; };

#define TB 256
#define BK 32
#define LDSPAD 4

// Problem constants
#define B_   8
#define M_   256
#define D_   1024
#define H_   16
#define HD_  64
#define L_   4096
#define LC_  4092
#define FF_  4096
#define KW_  5
#define R_   2048   // B_*M_
#define BH_  128    // B_*H_

// ---------------------------------------------------------------------------
// Generic batched WMMA GEMM with tap loop:
//   C[z] = alpha * sum_t A_t[z] @ op(B_t[z])  (+bias, +act)
// A: (M x K) bf16 row-major, lda; A_t = A + t*sAtap.
// B: (K x N) bf16 (or (N x K) if btrans); B_t = B + taps.o[t].
// Batch z = zb*hdiv + zh with independent (zb, zh) strides per matrix.
// Template: WR x WC wave grid (8 waves), each wave TI*16 x TJ*16 output.
// ---------------------------------------------------------------------------
template <int WR, int WC, int TI, int TJ>
__global__ __launch_bounds__(TB) void gemm_wmma_bf16(
    const bf16* __restrict__ A, long sAb, long sAh, int lda, int avec, long sAtap,
    const bf16* __restrict__ Bm, long sBb, long sBh, int ldb, int btrans, int bvec,
    Taps taps, int ntap,
    float* __restrict__ C, bf16* __restrict__ Cb, long sCb, long sCh, int ldc,
    const float* __restrict__ bias, int biasMode,   // 0 none, 1 col, 2 row
    int Mdim, int Ndim, int Kdim,
    float alpha, int act, int hdiv)
{
  constexpr int BMt = WR * TI * 16;
  constexpr int BNt = WC * TJ * 16;
  constexpr int NA4 = BMt * BK / (4 * TB);   // 8B chunks of A per thread
  constexpr int NB4 = BNt * BK / (4 * TB);   // 8B chunks of B per thread
  __shared__ alignas(16) bf16 As[BMt][BK + LDSPAD];
  __shared__ alignas(16) bf16 Bs[BNt][BK + LDSPAD];

  const int z  = blockIdx.z;
  const int zb = z / hdiv, zh = z % hdiv;
  A  += (long)zb * sAb + (long)zh * sAh;
  Bm += (long)zb * sBb + (long)zh * sBh;
  const long coff = (long)zb * sCb + (long)zh * sCh;

  const int tid  = threadIdx.x;
  const int wave = tid >> 5, lane = tid & 31;
  const int lr = lane & 15, hi = lane >> 4;
  const int wm = wave / WC, wn = wave % WC;
  const int rowBase = blockIdx.y * BMt;
  const int colBase = blockIdx.x * BNt;
  const int row0 = rowBase + wm * (TI * 16);
  const int col0 = colBase + wn * (TJ * 16);

  const bool fullA = (rowBase + BMt <= Mdim);
  const bool fullB = (colBase + BNt <= Ndim);

  Pack4 ra[NA4], rb[NB4];

  // Load one (tap, k-step) tile into registers (vector fast path / clamped
  // branch-free scalar path on edges).
  auto loadTile = [&](int t, int k0) {
    const bf16* At = A + (long)t * sAtap;
    const bf16* Bt = Bm + taps.o[t];
    const bool fullK = (k0 + BK <= Kdim);
    if (fullA && fullK && avec) {
#pragma unroll
      for (int j = 0; j < NA4; ++j) {
        int i = tid + j * TB;
        int r = i >> 3, c4 = (i & 7) * 4;
        ra[j].u = *reinterpret_cast<const unsigned long long*>(
            At + (long)(rowBase + r) * lda + (k0 + c4));
      }
    } else {
#pragma unroll
      for (int j = 0; j < NA4; ++j) {
        int i = tid + j * TB;
        int r = i >> 3, c4 = (i & 7) * 4;
        int gr = rowBase + r;
#pragma unroll
        for (int x = 0; x < 4; ++x) {
          int gk = k0 + c4 + x;
          bool ok = (gr < Mdim) && (gk < Kdim);
          long idx = ok ? ((long)gr * lda + gk) : 0;
          bf16 v = At[idx];
          ra[j].e[x] = ok ? v : (bf16)0.f;
        }
      }
    }
    if (fullB && fullK && bvec) {
      if (btrans) {
#pragma unroll
        for (int j = 0; j < NB4; ++j) {
          int i = tid + j * TB;
          int r = i >> 3, c4 = (i & 7) * 4;
          rb[j].u = *reinterpret_cast<const unsigned long long*>(
              Bt + (long)(colBase + r) * ldb + (k0 + c4));
        }
      } else {
#pragma unroll
        for (int j = 0; j < NB4; ++j) {
          int i = tid + j * TB;
          int kk = i / (BNt / 4), n4 = (i % (BNt / 4)) * 4;
          rb[j].u = *reinterpret_cast<const unsigned long long*>(
              Bt + (long)(k0 + kk) * ldb + (colBase + n4));
        }
      }
    } else {
      if (btrans) {
#pragma unroll
        for (int j = 0; j < NB4; ++j) {
          int i = tid + j * TB;
          int r = i >> 3, c4 = (i & 7) * 4;
          int gn = colBase + r;
#pragma unroll
          for (int x = 0; x < 4; ++x) {
            int gk = k0 + c4 + x;
            bool ok = (gn < Ndim) && (gk < Kdim);
            long idx = ok ? ((long)gn * ldb + gk) : 0;
            bf16 v = Bt[idx];
            rb[j].e[x] = ok ? v : (bf16)0.f;
          }
        }
      } else {
#pragma unroll
        for (int j = 0; j < NB4; ++j) {
          int i = tid + j * TB;
          int kk = i / (BNt / 4), n4 = (i % (BNt / 4)) * 4;
          int gk = k0 + kk;
#pragma unroll
          for (int x = 0; x < 4; ++x) {
            int gn = colBase + n4 + x;
            bool ok = (gn < Ndim) && (gk < Kdim);
            long idx = ok ? ((long)gk * ldb + gn) : 0;
            bf16 v = Bt[idx];
            rb[j].e[x] = ok ? v : (bf16)0.f;
          }
        }
      }
    }
  };

  // Dump register-staged tile to LDS.  A (and transposed B) rows are
  // K-contiguous -> 8B ds stores; non-transposed B scatters 4 b16 stores.
  auto storeTile = [&]() {
#pragma unroll
    for (int j = 0; j < NA4; ++j) {
      int i = tid + j * TB;
      int r = i >> 3, c4 = (i & 7) * 4;
      *reinterpret_cast<unsigned long long*>(&As[r][c4]) = ra[j].u;
    }
    if (btrans) {
#pragma unroll
      for (int j = 0; j < NB4; ++j) {
        int i = tid + j * TB;
        int r = i >> 3, c4 = (i & 7) * 4;
        *reinterpret_cast<unsigned long long*>(&Bs[r][c4]) = rb[j].u;
      }
    } else {
#pragma unroll
      for (int j = 0; j < NB4; ++j) {
        int i = tid + j * TB;
        int kk = i / (BNt / 4), n4 = (i % (BNt / 4)) * 4;
        Bs[n4 + 0][kk] = rb[j].e[0];
        Bs[n4 + 1][kk] = rb[j].e[1];
        Bs[n4 + 2][kk] = rb[j].e[2];
        Bs[n4 + 3][kk] = rb[j].e[3];
      }
    }
  };

  CFrag acc[TI][TJ];
#pragma unroll
  for (int ti = 0; ti < TI; ++ti)
#pragma unroll
    for (int tj = 0; tj < TJ; ++tj)
#pragma unroll
      for (int e = 0; e < 8; ++e) acc[ti][tj].e[e] = 0.f;

  const int ksteps = (Kdim + BK - 1) / BK;
  const int total  = ntap * ksteps;

  loadTile(0, 0);
  storeTile();
  __syncthreads();

  for (int it = 0; it < total; ++it) {
    const int nt = it + 1;
    const bool hasNext = nt < total;
    if (hasNext) loadTile(nt / ksteps, (nt % ksteps) * BK);  // prefetch to regs

    // ---- fragments per documented CDNA5 16-bit WMMA VGPR layouts ----
    AFrag af[TI], bfg[TJ];
#pragma unroll
    for (int ti = 0; ti < TI; ++ti) {
      const bf16* arow = &As[wm * (TI * 16) + ti * 16 + lr][0];
#pragma unroll
      for (int v = 0; v < 8; ++v) {
        int kk = ((v & 3) << 1) + (hi << 3) + ((v >> 2) << 4);
        af[ti].e[2 * v]     = arow[kk];
        af[ti].e[2 * v + 1] = arow[kk + 1];
      }
    }
#pragma unroll
    for (int tj = 0; tj < TJ; ++tj) {
      const bf16* bcol = &Bs[wn * (TJ * 16) + tj * 16 + lr][0];
#pragma unroll
      for (int v = 0; v < 8; ++v) {
        int kk = ((v & 3) << 1) + (hi << 3) + ((v >> 2) << 4);
        bfg[tj].e[2 * v]     = bcol[kk];
        bfg[tj].e[2 * v + 1] = bcol[kk + 1];
      }
    }
#pragma unroll
    for (int ti = 0; ti < TI; ++ti)
#pragma unroll
      for (int tj = 0; tj < TJ; ++tj)
        acc[ti][tj].v = __builtin_amdgcn_wmma_f32_16x16x32_bf16(
            false, af[ti].v, false, bfg[tj].v, (short)0, acc[ti][tj].v,
            false, false);

    __syncthreads();                // all waves done reading LDS
    if (hasNext) storeTile();       // regs -> LDS for next iteration
    __syncthreads();
  }

#pragma unroll
  for (int ti = 0; ti < TI; ++ti)
#pragma unroll
    for (int tj = 0; tj < TJ; ++tj)
#pragma unroll
      for (int e = 0; e < 8; ++e) {
        int r = row0 + ti * 16 + hi * 8 + e;
        int c = col0 + tj * 16 + lr;
        if (r < Mdim && c < Ndim) {
          float x = acc[ti][tj].e[e] * alpha;
          if (bias) x += (biasMode == 2) ? bias[r] : bias[c];
          if (act == 1) x = fmaxf(x, 0.f);
          long o = coff + (long)r * ldc + c;
          if (C)  C[o]  = x;
          if (Cb) Cb[o] = (bf16)x;
        }
      }
}

// ---------------------------------------------------------------------------
// Elementwise / reduction helper kernels (f32)
// ---------------------------------------------------------------------------
__global__ __launch_bounds__(TB) void cast_bf16_kernel(
    const float* __restrict__ in, bf16* __restrict__ out, long n) {
  long i = (long)blockIdx.x * TB + threadIdx.x;
  if (i < n) out[i] = (bf16)in[i];
}

// Row-local shifted cast: out[row][l] = in[row][l+p] (0 beyond row end)
__global__ __launch_bounds__(TB) void cast_shift_kernel(
    const float* __restrict__ in, bf16* __restrict__ out, long rows, int Lr, int p) {
  long i = (long)blockIdx.x * TB + threadIdx.x;
  if (i >= rows * (long)Lr) return;
  int l = (int)(i % Lr);
  out[i] = (l + p < Lr) ? (bf16)in[i + p] : (bf16)0.f;
}

// out is (C x R): out[c*R + r] = in[r*C + c]  (pre-transpose weights)
__global__ __launch_bounds__(TB) void cast_transpose_kernel(
    const float* __restrict__ in, bf16* __restrict__ out, int R, int Cc) {
  long i = (long)blockIdx.x * TB + threadIdx.x;
  if (i >= (long)R * Cc) return;
  int r = (int)(i / Cc), c = (int)(i % Cc);
  out[(long)c * R + r] = (bf16)in[i];
}

__global__ __launch_bounds__(TB) void gate_kernel(
    float* __restrict__ f, const float* __restrict__ g, long n) {
  long i = (long)blockIdx.x * TB + threadIdx.x;
  if (i >= n) return;
  float fv = f[i], gv = g[i];
  float sig = 1.f / (1.f + __expf(-gv));
  f[i] = (tanhf(fv) + 0.05f * fv) * sig;
}

__global__ __launch_bounds__(TB) void bn_stats_kernel(
    const float* __restrict__ hbuf, float* __restrict__ meanO,
    float* __restrict__ rstdO, int Bn, int Cn, int Ln) {
  __shared__ float sm[TB], sm2[TB];
  int c = blockIdx.x, tid = threadIdx.x;
  float s = 0.f, s2 = 0.f;
  for (int b = 0; b < Bn; ++b) {
    const float* p = hbuf + ((long)b * Cn + c) * Ln;
    for (int i = tid; i < Ln; i += TB) { float v = p[i]; s += v; s2 += v * v; }
  }
  sm[tid] = s; sm2[tid] = s2; __syncthreads();
  for (int st = TB / 2; st > 0; st >>= 1) {
    if (tid < st) { sm[tid] += sm[tid + st]; sm2[tid] += sm2[tid + st]; }
    __syncthreads();
  }
  if (tid == 0) {
    float n = (float)Bn * (float)Ln;
    float mu = sm[0] / n;
    float var = sm2[0] / n - mu * mu;
    meanO[c] = mu;
    rstdO[c] = rsqrtf(var + 1e-5f);
  }
}

__global__ __launch_bounds__(TB) void bn_apply_kernel(
    const float* __restrict__ hbuf, const float* __restrict__ mean,
    const float* __restrict__ rstd, const float* __restrict__ g,
    const float* __restrict__ bta, float* __restrict__ outF,
    bf16* __restrict__ outB, long total, int Cn, int Ln) {
  long i = (long)blockIdx.x * TB + threadIdx.x;
  if (i >= total) return;
  int c = (int)((i / Ln) % Cn);
  float v = (hbuf[i] - mean[c]) * rstd[c] * g[c] + bta[c];
  outF[i] = v;
  outB[i] = (bf16)v;
}

__global__ __launch_bounds__(TB) void softmax_rows_kernel(
    const float* __restrict__ S, bf16* __restrict__ P, int n) {
  __shared__ float sm[TB];
  long row = blockIdx.x;
  const float* s = S + row * (long)n;
  bf16* p = P + row * (long)n;
  int tid = threadIdx.x;
  float m = -3.4e38f;
  for (int i = tid; i < n; i += TB) m = fmaxf(m, s[i]);
  sm[tid] = m; __syncthreads();
  for (int st = TB / 2; st > 0; st >>= 1) {
    if (tid < st) sm[tid] = fmaxf(sm[tid], sm[tid + st]);
    __syncthreads();
  }
  m = sm[0]; __syncthreads();
  float sum = 0.f;
  for (int i = tid; i < n; i += TB) sum += __expf(s[i] - m);
  sm[tid] = sum; __syncthreads();
  for (int st = TB / 2; st > 0; st >>= 1) {
    if (tid < st) sm[tid] += sm[tid + st];
    __syncthreads();
  }
  float inv = 1.f / sm[0];
  for (int i = tid; i < n; i += TB) p[i] = (bf16)(__expf(s[i] - m) * inv);
}

// out = LayerNorm(resid + delta); optional bf16 mirror
__global__ __launch_bounds__(TB) void resid_ln_kernel(
    const float* __restrict__ resid, const float* __restrict__ delta,
    const float* __restrict__ g, const float* __restrict__ bta,
    float* __restrict__ out, bf16* __restrict__ outB, int n) {
  __shared__ float sm[TB];
  long row = blockIdx.x;
  const float* x = resid + row * (long)n;
  const float* y = delta + row * (long)n;
  int tid = threadIdx.x;
  float loc[8];
  int cnt = 0;
  float s = 0.f;
  for (int i = tid; i < n; i += TB) { float v = x[i] + y[i]; loc[cnt++] = v; s += v; }
  sm[tid] = s; __syncthreads();
  for (int st = TB / 2; st > 0; st >>= 1) {
    if (tid < st) sm[tid] += sm[tid + st];
    __syncthreads();
  }
  float mean = sm[0] / n; __syncthreads();
  float s2 = 0.f;
  for (int j = 0; j < cnt; ++j) { float d = loc[j] - mean; s2 += d * d; }
  sm[tid] = s2; __syncthreads();
  for (int st = TB / 2; st > 0; st >>= 1) {
    if (tid < st) sm[tid] += sm[tid + st];
    __syncthreads();
  }
  float rstd = rsqrtf(sm[0] / n + 1e-5f); __syncthreads();
  cnt = 0;
  for (int i = tid; i < n; i += TB) {
    float v = (loc[cnt++] - mean) * rstd * g[i] + bta[i];
    out[row * (long)n + i] = v;
    if (outB) outB[row * (long)n + i] = (bf16)v;
  }
}

// ---------------------------------------------------------------------------
// Host side
// ---------------------------------------------------------------------------
static inline void launch_gemm(hipStream_t st, bool narrowN,
    const bf16* A, long sAb, long sAh, int lda, long sAtap,
    const bf16* B, long sBb, long sBh, int ldb, int btrans, Taps taps, int ntap,
    float* C, bf16* Cb, long sCb, long sCh, int ldc,
    const float* bias, int biasMode,
    int M, int N, int K, float alpha, int act, int hdiv, int Z) {
  int avec = ((((uintptr_t)A) & 7) == 0) && (lda % 4 == 0);
  int bvec = ((((uintptr_t)B) & 7) == 0) && (ldb % 4 == 0);
  if (narrowN) {
    dim3 g((N + 63) / 64, (M + 63) / 64, Z);
    gemm_wmma_bf16<4, 2, 1, 2><<<g, dim3(TB), 0, st>>>(
        A, sAb, sAh, lda, avec, sAtap, B, sBb, sBh, ldb, btrans, bvec, taps, ntap,
        C, Cb, sCb, sCh, ldc, bias, biasMode, M, N, K, alpha, act, hdiv);
  } else {
    dim3 g((N + 127) / 128, (M + 127) / 128, Z);
    gemm_wmma_bf16<4, 2, 2, 4><<<g, dim3(TB), 0, st>>>(
        A, sAb, sAh, lda, avec, sAtap, B, sBb, sBh, ldb, btrans, bvec, taps, ntap,
        C, Cb, sCb, sCh, ldc, bias, biasMode, M, N, K, alpha, act, hdiv);
  }
}

extern "C" void kernel_launch(void* const* d_in, const int* in_sizes, int n_in,
                              void* d_out, int out_size, void* d_ws,
                              size_t ws_size, hipStream_t stream) {
  (void)in_sizes; (void)n_in; (void)out_size; (void)ws_size;
  const float* memory   = (const float*)d_in[0];
  const float* sequence = (const float*)d_in[1];
  const float* sattn_in_w  = (const float*)d_in[2];
  const float* sattn_in_b  = (const float*)d_in[3];
  const float* sattn_out_w = (const float*)d_in[4];
  const float* sattn_out_b = (const float*)d_in[5];
  const float* xq_w  = (const float*)d_in[6];
  const float* xk_wT = (const float*)d_in[7];
  const float* xv_w  = (const float*)d_in[8];
  const float* xq_b  = (const float*)d_in[9];
  const float* xk_b  = (const float*)d_in[10];
  const float* xv_b  = (const float*)d_in[11];
  const float* xo_w  = (const float*)d_in[12];
  const float* xo_b  = (const float*)d_in[13];
  const float* n1_g = (const float*)d_in[14];
  const float* n1_b = (const float*)d_in[15];
  const float* n2_g = (const float*)d_in[16];
  const float* n2_b = (const float*)d_in[17];
  const float* n3_g = (const float*)d_in[18];
  const float* n3_b = (const float*)d_in[19];
  const float* ff_w1 = (const float*)d_in[20];
  const float* ff_b1 = (const float*)d_in[21];
  const float* ff_w2 = (const float*)d_in[22];
  const float* ff_b2 = (const float*)d_in[23];
  const float* fc_w = (const float*)d_in[24];
  const float* fc_b = (const float*)d_in[25];
  const float* gc_w = (const float*)d_in[26];
  const float* gc_b = (const float*)d_in[27];
  const float* bn_g = (const float*)d_in[28];
  const float* bn_b = (const float*)d_in[29];

  float* out_mem = (float*)d_out;                       // (B, M, D)
  float* out_seq = (float*)d_out + (long)B_ * M_ * D_;  // (B, D, LC)

  // --- workspace bump allocator ---
  char* base = (char*)d_ws;
  size_t cur = 0;
  auto alloc = [&](long elems, int esz) -> void* {
    void* p = base + cur;
    cur = (cur + (size_t)elems * esz + 255) & ~(size_t)255;
    return p;
  };

  const long PH = (long)B_ * D_ * L_;        // one phase copy, elements
  bf16* seqph = (bf16*)alloc(4 * PH, 2);     // 4 contiguous phase-shifted copies
  bf16* membf  = (bf16*)alloc((long)R_ * D_, 2);
  bf16* wqkvT  = (bf16*)alloc((long)D_ * 3 * D_, 2);
  bf16* wsoutT = (bf16*)alloc((long)D_ * D_, 2);
  bf16* wxq    = (bf16*)alloc((long)D_ * D_, 2);
  bf16* wxk    = (bf16*)alloc((long)D_ * D_, 2);
  bf16* wxvT   = (bf16*)alloc((long)D_ * D_, 2);
  bf16* wxoT   = (bf16*)alloc((long)D_ * D_, 2);
  bf16* wff1T  = (bf16*)alloc((long)D_ * FF_, 2);
  bf16* wff2T  = (bf16*)alloc((long)FF_ * D_, 2);
  bf16* wfck   = (bf16*)alloc((long)KW_ * D_ * D_, 2);
  bf16* wgck   = (bf16*)alloc((long)KW_ * D_ * D_, 2);
  bf16* qkvbf  = (bf16*)alloc((long)R_ * 3 * D_, 2);
  float* sself = (float*)alloc((long)BH_ * M_ * M_, 4);
  bf16* pself  = (bf16*)alloc((long)BH_ * M_ * M_, 2);
  bf16* xaself = (bf16*)alloc((long)R_ * D_, 2);
  float* saout = (float*)alloc((long)R_ * D_, 4);
  float* mem1f = (float*)alloc((long)R_ * D_, 4);
  bf16* mem1b  = (bf16*)alloc((long)R_ * D_, 2);
  bf16* qxbf   = (bf16*)alloc((long)R_ * D_, 2);
  float* fcx   = (float*)alloc((long)B_ * D_ * LC_, 4);
  float* gcx   = (float*)alloc((long)B_ * D_ * LC_, 4);
  float* bnmean = (float*)alloc(D_, 4);
  float* bnrstd = (float*)alloc(D_, 4);
  bf16* seqnb  = (bf16*)alloc((long)B_ * D_ * LC_, 2);
  bf16* ktb    = (bf16*)alloc((long)B_ * D_ * LC_, 2);
  bf16* vtb    = (bf16*)alloc((long)B_ * D_ * LC_, 2);
  float* scross = (float*)alloc((long)BH_ * M_ * LC_, 4);
  bf16* pcross  = (bf16*)alloc((long)BH_ * M_ * LC_, 2);
  bf16* xaxbf  = (bf16*)alloc((long)R_ * D_, 2);
  float* xoout = (float*)alloc((long)R_ * D_, 4);
  float* mem2f = (float*)alloc((long)R_ * D_, 4);
  bf16* mem2b  = (bf16*)alloc((long)R_ * D_, 2);
  bf16* ffhbf  = (bf16*)alloc((long)R_ * FF_, 2);
  float* ffof  = (float*)alloc((long)R_ * D_, 4);

  auto gridFor = [](long n) { return (unsigned)((n + TB - 1) / TB); };
  Taps tz{};   // zero taps for ordinary GEMMs

  // --- casts / weight pre-packs ---
  for (int p = 0; p < 4; ++p)
    cast_shift_kernel<<<gridFor(PH), TB, 0, stream>>>(
        sequence, seqph + p * PH, (long)B_ * D_, L_, p);
  cast_bf16_kernel<<<gridFor((long)R_ * D_), TB, 0, stream>>>(memory, membf, (long)R_ * D_);
  cast_transpose_kernel<<<gridFor((long)3 * D_ * D_), TB, 0, stream>>>(sattn_in_w, wqkvT, 3 * D_, D_);
  cast_transpose_kernel<<<gridFor((long)D_ * D_), TB, 0, stream>>>(sattn_out_w, wsoutT, D_, D_);
  cast_bf16_kernel<<<gridFor((long)D_ * D_), TB, 0, stream>>>(xq_w, wxq, (long)D_ * D_);
  cast_bf16_kernel<<<gridFor((long)D_ * D_), TB, 0, stream>>>(xk_wT, wxk, (long)D_ * D_);
  cast_transpose_kernel<<<gridFor((long)D_ * D_), TB, 0, stream>>>(xv_w, wxvT, D_, D_);
  cast_transpose_kernel<<<gridFor((long)D_ * D_), TB, 0, stream>>>(xo_w, wxoT, D_, D_);
  cast_transpose_kernel<<<gridFor((long)FF_ * D_), TB, 0, stream>>>(ff_w1, wff1T, FF_, D_);
  cast_transpose_kernel<<<gridFor((long)D_ * FF_), TB, 0, stream>>>(ff_w2, wff2T, D_, FF_);
  cast_transpose_kernel<<<gridFor((long)D_ * D_ * KW_), TB, 0, stream>>>(fc_w, wfck, D_ * D_, KW_);
  cast_transpose_kernel<<<gridFor((long)D_ * D_ * KW_), TB, 0, stream>>>(gc_w, wgck, D_ * D_, KW_);

  // --- gated conv: single GEMM per branch, 5 taps folded in-kernel ---
  Taps tconv{};
  for (int k = 0; k < KW_; ++k)
    tconv.o[k] = (long)(k & 3) * PH + (k & ~3);   // 8B-aligned tap offsets
  launch_gemm(stream, false, wfck, 0, 0, D_, (long)D_ * D_,
              seqph, (long)D_ * L_, 0, L_, 0, tconv, KW_,
              fcx, nullptr, (long)D_ * LC_, 0, LC_, fc_b, 2,
              D_, LC_, D_, 1.f, 0, 1, B_);
  launch_gemm(stream, false, wgck, 0, 0, D_, (long)D_ * D_,
              seqph, (long)D_ * L_, 0, L_, 0, tconv, KW_,
              gcx, nullptr, (long)D_ * LC_, 0, LC_, gc_b, 2,
              D_, LC_, D_, 1.f, 0, 1, B_);
  long seqTotal = (long)B_ * D_ * LC_;
  gate_kernel<<<gridFor(seqTotal), TB, 0, stream>>>(fcx, gcx, seqTotal);
  bn_stats_kernel<<<D_, TB, 0, stream>>>(fcx, bnmean, bnrstd, B_, D_, LC_);
  bn_apply_kernel<<<gridFor(seqTotal), TB, 0, stream>>>(
      fcx, bnmean, bnrstd, bn_g, bn_b, out_seq, seqnb, seqTotal, D_, LC_);

  // --- self-attention ---
  launch_gemm(stream, false, membf, 0, 0, D_, 0, wqkvT, 0, 0, 3 * D_, 0, tz, 1,
              nullptr, qkvbf, 0, 0, 3 * D_, sattn_in_b, 1,
              R_, 3 * D_, D_, 1.f, 0, 1, 1);
  launch_gemm(stream, false, qkvbf, (long)M_ * 3 * D_, HD_, 3 * D_, 0,
              qkvbf + D_, (long)M_ * 3 * D_, HD_, 3 * D_, 1, tz, 1,
              sself, nullptr, (long)H_ * M_ * M_, (long)M_ * M_, M_,
              nullptr, 0, M_, M_, HD_, 0.125f, 0, H_, BH_);
  softmax_rows_kernel<<<BH_ * M_, TB, 0, stream>>>(sself, pself, M_);
  launch_gemm(stream, true, pself, (long)H_ * M_ * M_, (long)M_ * M_, M_, 0,
              qkvbf + 2 * D_, (long)M_ * 3 * D_, HD_, 3 * D_, 0, tz, 1,
              nullptr, xaself, (long)M_ * D_, HD_, D_,
              nullptr, 0, M_, HD_, M_, 1.f, 0, H_, BH_);
  launch_gemm(stream, false, xaself, 0, 0, D_, 0, wsoutT, 0, 0, D_, 0, tz, 1,
              saout, nullptr, 0, 0, D_, sattn_out_b, 1,
              R_, D_, D_, 1.f, 0, 1, 1);
  resid_ln_kernel<<<R_, TB, 0, stream>>>(memory, saout, n1_g, n1_b, mem1f, mem1b, D_);

  // --- cross-attention ---
  launch_gemm(stream, false, mem1b, 0, 0, D_, 0, wxq, 0, 0, D_, 0, tz, 1,
              nullptr, qxbf, 0, 0, D_, xq_b, 1,
              R_, D_, D_, 1.f, 0, 1, 1);
  launch_gemm(stream, false, wxk, 0, 0, D_, 0, seqnb, (long)D_ * LC_, 0, LC_, 0, tz, 1,
              nullptr, ktb, (long)D_ * LC_, 0, LC_, xk_b, 2,
              D_, LC_, D_, 1.f, 0, 1, B_);
  launch_gemm(stream, false, wxvT, 0, 0, D_, 0, seqnb, (long)D_ * LC_, 0, LC_, 0, tz, 1,
              nullptr, vtb, (long)D_ * LC_, 0, LC_, xv_b, 2,
              D_, LC_, D_, 1.f, 0, 1, B_);
  launch_gemm(stream, false, qxbf, (long)M_ * D_, HD_, D_, 0,
              ktb, (long)D_ * LC_, (long)HD_ * LC_, LC_, 0, tz, 1,
              scross, nullptr, (long)H_ * M_ * LC_, (long)M_ * LC_, LC_,
              nullptr, 0, M_, LC_, HD_, 0.03125f, 0, H_, BH_);
  softmax_rows_kernel<<<BH_ * M_, TB, 0, stream>>>(scross, pcross, LC_);
  launch_gemm(stream, true, pcross, (long)H_ * M_ * LC_, (long)M_ * LC_, LC_, 0,
              vtb, (long)D_ * LC_, (long)HD_ * LC_, LC_, 1, tz, 1,
              nullptr, xaxbf, (long)M_ * D_, HD_, D_,
              nullptr, 0, M_, HD_, LC_, 1.f, 0, H_, BH_);
  launch_gemm(stream, false, xaxbf, 0, 0, D_, 0, wxoT, 0, 0, D_, 0, tz, 1,
              xoout, nullptr, 0, 0, D_, xo_b, 1,
              R_, D_, D_, 1.f, 0, 1, 1);
  resid_ln_kernel<<<R_, TB, 0, stream>>>(mem1f, xoout, n2_g, n2_b, mem2f, mem2b, D_);

  // --- FFN ---
  launch_gemm(stream, false, mem2b, 0, 0, D_, 0, wff1T, 0, 0, FF_, 0, tz, 1,
              nullptr, ffhbf, 0, 0, FF_, ff_b1, 1,
              R_, FF_, D_, 1.f, 1, 1, 1);
  launch_gemm(stream, false, ffhbf, 0, 0, FF_, 0, wff2T, 0, 0, D_, 0, tz, 1,
              ffof, nullptr, 0, 0, D_, ff_b2, 1,
              R_, D_, FF_, 1.f, 0, 1, 1);
  resid_ln_kernel<<<R_, TB, 0, stream>>>(mem2f, ffof, n3_g, n3_b, out_mem, nullptr, D_);
}
